// LSTMRecommender_57045755626073
// MI455X (gfx1250) — matrix operand order repaired and checked by reading, not compiled, not run on hardware
//
#include <hip/hip_runtime.h>
#include <stdint.h>

// ---------------- types ----------------
typedef __attribute__((ext_vector_type(16))) __bf16 v16bf;
typedef __attribute__((ext_vector_type(8)))  float  v8f;

struct alignas(16) U4 { unsigned int a, b, c, d; };
union FragU { U4 q[2]; v16bf v; };

#define B_SZ   1024
#define S_LEN  50
#define L_LEN  20
#define P_DIM  64
#define C_DIM  32
#define HID    128
#define GATES  512
#define KCAT   256
#define NPROD  100001

// ---------------- helpers ----------------
__device__ __forceinline__ unsigned short f2bf(float f) {
    union { float f; unsigned u; } v; v.f = f;
    unsigned r = v.u + 0x7FFFu + ((v.u >> 16) & 1u);
    return (unsigned short)(r >> 16);
}
__device__ __forceinline__ float bf2f(unsigned short h) {
    union { unsigned u; float f; } v; v.u = ((unsigned)h) << 16;
    return v.f;
}
__device__ __forceinline__ float sigm(float x) { return 1.0f / (1.0f + __expf(-x)); }

// A fragment (16x32 bf16, M x K) from row-major [row][ldk] bf16.
// lane: m = lane%16; 16 contiguous K at k0 + 16*(lane>=16).
__device__ __forceinline__ v16bf load_frag_a(const unsigned short* base, int row,
                                             int ldk, int k0, int lane) {
    int hi = lane >> 4;
    const unsigned short* p = base + (size_t)row * ldk + k0 + hi * 16;
    FragU u;
    u.q[0] = *(const U4*)(p);
    u.q[1] = *(const U4*)(p + 8);
    return u.v;
}

// B fragment (32x16 bf16, K x N) where B[k][n] = W[n][k], W row-major [n][ldk].
// lane: n = lane%16; 16 contiguous K at k0 + 16*(lane>=16).
__device__ __forceinline__ v16bf load_frag_b(const unsigned short* base, int n,
                                             int ldk, int k0, int lane) {
    int hi = lane >> 4;
    const unsigned short* p = base + (size_t)n * ldk + k0 + hi * 16;
    FragU u;
    u.q[0] = *(const U4*)(p);
    u.q[1] = *(const U4*)(p + 8);
    return u.v;
}

// ---------------- prep kernels ----------------
__global__ void k_f32_to_bf16(const float* __restrict__ w,
                              unsigned short* __restrict__ o, int n) {
    int i = blockIdx.x * blockDim.x + threadIdx.x;
    if (i < n) o[i] = f2bf(w[i]);
}

// build concatenated [512][256] bf16 weight: cols 0..127 = W_ih, 128..255 = W_hh
__global__ void k_wcat(const float* __restrict__ Wih, const float* __restrict__ Whh,
                       unsigned short* __restrict__ out) {
    int idx = blockIdx.x * blockDim.x + threadIdx.x;   // 512*256
    int n = idx >> 8, k = idx & 255;
    float v = (k < HID) ? Wih[n * HID + k] : Whh[n * HID + (k - HID)];
    out[idx] = f2bf(v);
}

__global__ void k_bias(const float* __restrict__ a, const float* __restrict__ b,
                       float* __restrict__ o) {
    int i = blockIdx.x * blockDim.x + threadIdx.x;
    if (i < GATES) o[i] = a[i] + b[i];
}

// ---------------- feature build: gather+mean pool + small linears ----------------
// One wave per (b,s). Output X time-major [S][B][128] bf16.
__global__ void __launch_bounds__(256) k_features(
    const int* __restrict__ prod_idx, const int* __restrict__ cat_idx,
    const float* __restrict__ age, const float* __restrict__ ts_in,
    const float* __restrict__ gender,
    const float* __restrict__ emb_p, const float* __restrict__ emb_c,
    const float* __restrict__ W_ts, const float* __restrict__ b_ts,
    const float* __restrict__ W_uf, const float* __restrict__ b_uf,
    unsigned short* __restrict__ X) {
    int w = blockIdx.x * 8 + (threadIdx.x >> 5);       // 51200 waves exactly
    int lane = threadIdx.x & 31;
    int b = w / S_LEN, s = w % S_LEN;
    const int* pr = prod_idx + (size_t)(b * S_LEN + s) * L_LEN;
    const int* cr = cat_idx  + (size_t)(b * S_LEN + s) * L_LEN;
    float p0 = 0.f, p1 = 0.f, cs = 0.f;
    for (int l = 0; l < L_LEN; ++l) {
        int pi = pr[l];
        const float* e = emb_p + (size_t)pi * P_DIM + 2 * lane;
        p0 += e[0];
        p1 += e[1];
        int ci = cr[l];
        cs += emb_c[(size_t)ci * C_DIM + lane];
    }
    const float inv = 1.0f / (float)L_LEN;
    p0 *= inv; p1 *= inv; cs *= inv;
    unsigned short* xr = X + ((size_t)s * B_SZ + b) * 128;
    xr[2 * lane]     = f2bf(p0);
    xr[2 * lane + 1] = f2bf(p1);
    xr[64 + lane]    = f2bf(cs);
    if (lane < 16) {
        float t = ts_in[b * S_LEN + s];
        xr[96 + lane]  = f2bf(W_ts[lane] * t + b_ts[lane]);
        float uf = W_uf[lane * 2] * age[b] + W_uf[lane * 2 + 1] * gender[b] + b_uf[lane];
        xr[112 + lane] = f2bf(uf);
    }
}

// ---------------- persistent LSTM layer ----------------
// One WG owns 16 batch rows for ALL 50 steps of one layer.
// Weights [512][256] bf16 loaded into LDS once (256 KB); h tile lives in LDS (4 KB);
// cell state c lives in registers (8 floats/thread). Per step:
//   gates[16,512] = [x_t | h_{t-1}][16,256] @ Wcat.T  (bf16 WMMA, f32 acc via LDS staging)
// Per-step h is streamed to hseq_out for the next layer.
// Grid: 64 WGs, 256 threads (8 waves; wave w owns gate cols [64w, 64w+64)).
__global__ void __launch_bounds__(256) k_lstm_layer(
    const unsigned short* __restrict__ Wcat,     // [512][256] bf16
    const float* __restrict__ bias,              // [512]  (b_ih + b_hh)
    const unsigned short* __restrict__ xseq,     // [50][1024][128] bf16 input sequence
    unsigned short* __restrict__ hseq_out) {     // [50][1024][128] bf16 output sequence
    __shared__ unsigned short wlds[GATES * KCAT];   // 262144 B
    __shared__ float glds[16 * GATES];              //  32768 B
    __shared__ unsigned short hlds[16 * HID];       //   4096 B  -> 299008 B total

    const int tid  = threadIdx.x;
    const int lane = tid & 31;
    const int wave = tid >> 5;
    const int hi   = lane >> 4;
    const int nl   = lane & 15;
    const int rowbase = blockIdx.x * 16;
    const int row = rowbase + nl;

    // one-time cooperative weight fill (16384 x 16B) + h tile zero
    {
        const U4* src = (const U4*)Wcat;
        U4* dst = (U4*)wlds;
        for (int i = tid; i < (GATES * KCAT) / 8; i += 256) dst[i] = src[i];
        unsigned int* hz = (unsigned int*)hlds;
        for (int i = tid; i < (16 * HID) / 2; i += 256) hz[i] = 0u;
    }
    float creg[8];
#pragma unroll
    for (int t = 0; t < 8; ++t) creg[t] = 0.f;
    __syncthreads();

    for (int s = 0; s < S_LEN; ++s) {
        const unsigned short* xs = xseq + (size_t)s * B_SZ * HID;

        // A fragments: K = [x(128) | h(128)];  h read from LDS (local row = nl)
        v16bf afrag[8];
#pragma unroll
        for (int kf = 0; kf < 4; ++kf) afrag[kf]     = load_frag_a(xs, row, HID, kf * 32, lane);
#pragma unroll
        for (int kf = 0; kf < 4; ++kf) afrag[4 + kf] = load_frag_a(hlds, nl, HID, kf * 32, lane);

        // wave w -> gate columns [w*64, w*64+64)
#pragma unroll
        for (int nt = 0; nt < 4; ++nt) {
            const int n = wave * 64 + nt * 16 + nl;
            v8f acc = {};
#pragma unroll
            for (int kf = 0; kf < 8; ++kf) {
                FragU u;
                const unsigned short* p = wlds + (size_t)n * KCAT + kf * 32 + hi * 16;
                u.q[0] = *(const U4*)(p);
                u.q[1] = *(const U4*)(p + 8);
                acc = __builtin_amdgcn_wmma_f32_16x16x32_bf16(
                    false, afrag[kf], false, u.v, (short)0, acc, false, false);
            }
#pragma unroll
            for (int j = 0; j < 8; ++j) glds[(j + 8 * hi) * GATES + n] = acc[j];
        }
        __syncthreads();

        // elementwise LSTM cell: 16 rows x 128 units, c in registers
        unsigned short* hs = hseq_out + (size_t)s * B_SZ * HID;
#pragma unroll
        for (int t = 0; t < 8; ++t) {
            int idx = tid + t * 256;
            int m = idx >> 7;
            int j = idx & 127;
            float gi = glds[m * GATES + j]           + bias[j];
            float gf = glds[m * GATES + HID + j]     + bias[HID + j];
            float gg = glds[m * GATES + 2 * HID + j] + bias[2 * HID + j];
            float go = glds[m * GATES + 3 * HID + j] + bias[3 * HID + j];
            float cn = sigm(gf) * creg[t] + sigm(gi) * tanhf(gg);
            float hn = sigm(go) * tanhf(cn);
            creg[t] = cn;
            unsigned short hb = f2bf(hn);
            hlds[m * HID + j] = hb;
            hs[(size_t)(rowbase + m) * HID + j] = hb;
        }
        __syncthreads();
    }
}

// ---------------- head ----------------
__global__ void __launch_bounds__(256) k_head1(
    const unsigned short* __restrict__ h1, const float* __restrict__ W1,
    const float* __restrict__ b1, unsigned short* __restrict__ hidden) {
    int idx = blockIdx.x * blockDim.x + threadIdx.x;   // 1024*64
    int b = idx >> 6, o = idx & 63;
    float s = b1[o];
    const unsigned short* hr = h1 + (size_t)b * HID;
    const float* wr = W1 + (size_t)o * HID;
    for (int k = 0; k < HID; ++k) s += bf2f(hr[k]) * wr[k];
    hidden[idx] = f2bf(s < 0.f ? 0.f : s);
}

// logits[1024,100001] = hidden[1024,64] @ W2bf.T + b2   (K=64, 2 WMMA per tile)
// grid (391, 64): x = N-tile of 256 (8 waves x 32 cols), y = M-tile of 16.
__global__ void __launch_bounds__(256) k_head2(
    const unsigned short* __restrict__ hidden,  // [1024][64] bf16
    const unsigned short* __restrict__ W2bf,    // [100001][64] bf16
    const float* __restrict__ b2,               // [100001]
    float* __restrict__ out) {                  // [1024][100001]
    const int tid = threadIdx.x, lane = tid & 31, wave = tid >> 5;
    const int hi = lane >> 4, nl = lane & 15;
    const int mbase = blockIdx.y * 16;
    const int nbase = blockIdx.x * 256 + wave * 32;
    const int row = mbase + nl;
    v16bf a0 = load_frag_a(hidden, row, 64, 0,  lane);
    v16bf a1 = load_frag_a(hidden, row, 64, 32, lane);
#pragma unroll
    for (int nt = 0; nt < 2; ++nt) {
        int n = nbase + nt * 16 + nl;
        int nc = (n < NPROD) ? n : (NPROD - 1);
        v16bf b0 = load_frag_b(W2bf, nc, 64, 0,  lane);
        v16bf b1v = load_frag_b(W2bf, nc, 64, 32, lane);
        v8f acc = {};
        acc = __builtin_amdgcn_wmma_f32_16x16x32_bf16(false, a0, false, b0,  (short)0, acc, false, false);
        acc = __builtin_amdgcn_wmma_f32_16x16x32_bf16(false, a1, false, b1v, (short)0, acc, false, false);
        if (n < NPROD) {
            float bb = b2[n];
#pragma unroll
            for (int j = 0; j < 8; ++j)
                out[(size_t)(mbase + j + 8 * hi) * NPROD + n] = acc[j] + bb;
        }
    }
}

// ---------------- launch ----------------
extern "C" void kernel_launch(void* const* d_in, const int* in_sizes, int n_in,
                              void* d_out, int out_size, void* d_ws, size_t ws_size,
                              hipStream_t stream) {
    (void)in_sizes; (void)n_in; (void)out_size; (void)ws_size;
    const int*   prod  = (const int*)d_in[0];
    const int*   cat   = (const int*)d_in[1];
    const float* age   = (const float*)d_in[2];
    const float* tsin  = (const float*)d_in[3];
    const float* gen   = (const float*)d_in[4];
    const float* emb_p = (const float*)d_in[5];
    const float* emb_c = (const float*)d_in[6];
    const float* W_ts  = (const float*)d_in[7];
    const float* b_ts  = (const float*)d_in[8];
    const float* W_uf  = (const float*)d_in[9];
    const float* b_uf  = (const float*)d_in[10];
    const float* W_ih0 = (const float*)d_in[11];
    const float* W_hh0 = (const float*)d_in[12];
    const float* b_ih0 = (const float*)d_in[13];
    const float* b_hh0 = (const float*)d_in[14];
    const float* W_ih1 = (const float*)d_in[15];
    const float* W_hh1 = (const float*)d_in[16];
    const float* b_ih1 = (const float*)d_in[17];
    const float* b_hh1 = (const float*)d_in[18];
    const float* W1    = (const float*)d_in[19];
    const float* b1    = (const float*)d_in[20];
    const float* W2    = (const float*)d_in[21];
    const float* b2    = (const float*)d_in[22];
    float* out = (float*)d_out;

    char* ws = (char*)d_ws;
    const size_t SEQ_BYTES = (size_t)S_LEN * B_SZ * HID * 2;   // 13107200
    const size_t OFF_X     = 0;
    const size_t OFF_H0    = OFF_X + SEQ_BYTES;
    const size_t OFF_H1    = OFF_H0 + SEQ_BYTES;
    const size_t OFF_WC0   = OFF_H1 + SEQ_BYTES;                // 262144
    const size_t OFF_WC1   = OFF_WC0 + 262144;                  // 262144
    const size_t OFF_B0    = OFF_WC1 + 262144;                  // 2048
    const size_t OFF_B1    = OFF_B0 + 2048;                     // 2048
    const size_t OFF_HID   = OFF_B1 + 2048;                     // 131072
    const size_t OFF_W2    = OFF_HID + 131072;                  // 12800128

    unsigned short* X     = (unsigned short*)(ws + OFF_X);
    unsigned short* hseq0 = (unsigned short*)(ws + OFF_H0);
    unsigned short* hseq1 = (unsigned short*)(ws + OFF_H1);
    unsigned short* wc0   = (unsigned short*)(ws + OFF_WC0);
    unsigned short* wc1   = (unsigned short*)(ws + OFF_WC1);
    float*          bias0 = (float*)(ws + OFF_B0);
    float*          bias1 = (float*)(ws + OFF_B1);
    unsigned short* hid   = (unsigned short*)(ws + OFF_HID);
    unsigned short* w2bf  = (unsigned short*)(ws + OFF_W2);

    // weight conversions
    k_f32_to_bf16<<<25001, 256, 0, stream>>>(W2, w2bf, NPROD * 64);
    k_wcat<<<512, 256, 0, stream>>>(W_ih0, W_hh0, wc0);
    k_wcat<<<512, 256, 0, stream>>>(W_ih1, W_hh1, wc1);
    k_bias<<<2, 256, 0, stream>>>(b_ih0, b_hh0, bias0);
    k_bias<<<2, 256, 0, stream>>>(b_ih1, b_hh1, bias1);
    // features
    k_features<<<6400, 256, 0, stream>>>(prod, cat, age, tsin, gen, emb_p, emb_c,
                                         W_ts, b_ts, W_uf, b_uf, X);
    // persistent recurrence: one launch per layer, weights LDS-resident for all 50 steps
    k_lstm_layer<<<64, 256, 0, stream>>>(wc0, bias0, X, hseq0);
    k_lstm_layer<<<64, 256, 0, stream>>>(wc1, bias1, hseq0, hseq1);
    // head (final h of layer 1 = last time slice of hseq1)
    const unsigned short* h1last = hseq1 + (size_t)(S_LEN - 1) * B_SZ * HID;
    k_head1<<<256, 256, 0, stream>>>(h1last, W1, b1, hid);
    k_head2<<<dim3(391, 64), 256, 0, stream>>>(hid, w2bf, b2, out);
}
